// New_Gloabl_sum_embedding_90177133347636
// MI455X (gfx1250) — compile-verified, matrix operand order checked by reading, not run on hardware
//
#include <hip/hip_runtime.h>
#include <hip/hip_bf16.h>

typedef __attribute__((ext_vector_type(16))) _Float16 v16h;
typedef __attribute__((ext_vector_type(8)))  float    v8f;
typedef __attribute__((ext_vector_type(4)))  int      v4i;

// address-space-qualified views of a 16B vector for the async-to-LDS builtin
typedef __attribute__((address_space(1))) v4i av4i_g;   // global
typedef __attribute__((address_space(3))) v4i av4i_s;   // LDS

#define NB     512      // batch
#define DD     128      // embedding dim
#define SS     50       // tweets per user
#define TT     32       // words per tweet
#define LL     64       // latent
#define CC     50       // friend clip
#define NFR    10000    // NUM_FRIENDS (pad row id)
#define TSTR   136      // row-major tile stride (f16): 272B rows, 16B aligned, bank-clean
#define TSTRT  40       // transposed tile stride (f16): 80B rows, 16B aligned, bank-clean

#if defined(__has_builtin)
#  if __has_builtin(__builtin_amdgcn_global_load_async_to_lds_b128)
#    define HAS_ASYNC 1
#  else
#    define HAS_ASYNC 0
#  endif
#else
#  define HAS_ASYNC 0
#endif

union Frag16 { v16h h; v4i q[2]; };

__device__ __forceinline__ void async_b128(const float* g, float* l) {
#if HAS_ASYNC
    __builtin_amdgcn_global_load_async_to_lds_b128(
        (av4i_g*)(void*)g,       // global src (AS1, v4i*)
        (av4i_s*)(void*)l,       // LDS dst   (AS3, v4i*)
        0, 0);
#else
    (void)g; (void)l;
#endif
}

__device__ __forceinline__ void wait_async4() {
#if HAS_ASYNC
#  if __has_builtin(__builtin_amdgcn_s_wait_asynccnt)
    __builtin_amdgcn_s_wait_asynccnt(4);
#  else
    asm volatile("s_wait_asynccnt 0x4" ::: "memory");
#  endif
#endif
}
__device__ __forceinline__ void wait_async0() {
#if HAS_ASYNC
#  if __has_builtin(__builtin_amdgcn_s_wait_asynccnt)
    __builtin_amdgcn_s_wait_asynccnt(0);
#  else
    asm volatile("s_wait_asynccnt 0x0" ::: "memory");
#  endif
#endif
}

__launch_bounds__(256, 1)
__global__ void somn_kernel(
    const int*   __restrict__ user_idx,   // [B]
    const int*   __restrict__ item_idx,   // [B]
    const float* __restrict__ emb_word,   // [(PAD+1), D]
    const float* __restrict__ emb_item,   // [NUM_ITEMS, L]
    const float* __restrict__ emb_user,   // [NFR+1, L]
    const float* __restrict__ W_mem,      // [L, D]
    const float* __restrict__ b_mem,      // [D]
    const float* __restrict__ w_word,     // [D]
    const float* __restrict__ b_word,     // [1]
    const float* __restrict__ w_sent,     // [D]
    const float* __restrict__ b_sent,     // [1]
    const float* __restrict__ W_tr,       // [D, L]
    const float* __restrict__ b_tr,       // [L]
    const float* __restrict__ W1,         // [L, 32]
    const float* __restrict__ b1,         // [32]
    const float* __restrict__ W2,         // [32, 16]
    const float* __restrict__ b2,         // [16]
    const float* __restrict__ W3,         // [16, 2]
    const float* __restrict__ b3,         // [2]
    const float* __restrict__ w_aff,      // [L]
    const float* __restrict__ b_aff,      // [1]
    const int*   __restrict__ uti,        // [NUM_USERS, S, T]
    const int*   __restrict__ ufi,        // [NUM_USERS, C]
    float*       __restrict__ out)        // [B] rating ++ [B*C] group_idx
{
    __shared__ __align__(16) float    buf[2][TT * DD];     // f32 async staging ring (2x16KB)
    __shared__ __align__(16) _Float16 tile [TT * TSTR];    // 32x128 tweet tile, row major
    __shared__ __align__(16) _Float16 tileT[DD * TSTRT];   // same tile, transposed (d-major)
    __shared__ float    sentence[SS * DD];
    __shared__ float    memv[DD];
    __shared__ __align__(16) _Float16 memh[DD];
    __shared__ __align__(16) _Float16 wwh[TT];
    __shared__ __align__(16) float    scores[TT];
    __shared__ float    swv[SS];
    __shared__ float    item_s[LL];
    __shared__ float    ept_s[LL];
    __shared__ float    gi_s[CC];
    __shared__ int      fidx_s[CC];
    __shared__ float    a1s[32], a2s[16], alpha_s[2];
    __shared__ float    misc[2];              // [0]=iw, [1]=friend_num

    const int tid  = threadIdx.x;
    const int lane = tid & 31;
    const int wave = tid >> 5;
    const int b    = blockIdx.x;

    const int u  = user_idx[b];
    const int it = item_idx[b];

    // per-thread gather coordinates (thread -> 16 consecutive floats of one tile row)
    const int gt = tid >> 3;              // row t in 0..31
    const int gd = (tid & 7) * 16;        // d0 in 0..112

    if (tid < LL) item_s[tid] = emb_item[(long)it * LL + tid];
    __syncthreads();

    // mem = item_emb @ W_mem + b_mem
    if (tid < DD) {
        float acc = b_mem[tid];
        for (int l = 0; l < LL; ++l) acc += item_s[l] * W_mem[l * DD + tid];
        memv[tid] = acc;
    }
    __syncthreads();

    const int* uti_b = uti + (long)u * (SS * TT);

    for (int hoop = 0; hoop < 2; ++hoop) {
        if (tid < DD) memh[tid] = (_Float16)memv[tid];

#if HAS_ASYNC
        // prefetch tile s=0 into staging buffer 0 (4 async b128 per thread)
        {
            const int w = uti_b[0 * TT + gt];
            const float* g = emb_word + (long)w * DD + gd;
            float* l = &buf[0][gt * DD + gd];
            #pragma unroll
            for (int i = 0; i < 4; ++i) async_b128(g + 4 * i, l + 4 * i);
        }
#endif
        __syncthreads();

        for (int s = 0; s < SS; ++s) {
            union { _Float16 h[16]; v4i q[2]; } pk;
#if HAS_ASYNC
            // ---- issue async gather for tile s+1, retire tile s (in-order completion)
            if (s + 1 < SS) {
                const int w = uti_b[(s + 1) * TT + gt];
                const float* g = emb_word + (long)w * DD + gd;
                float* l = &buf[(s + 1) & 1][gt * DD + gd];
                #pragma unroll
                for (int i = 0; i < 4; ++i) async_b128(g + 4 * i, l + 4 * i);
                wait_async4();            // oldest 4 (tile s) complete
            } else {
                wait_async0();
            }
            __syncthreads();              // tile s fully resident in buf[s&1]
            {
                const float4* src = (const float4*)(&buf[s & 1][gt * DD + gd]);
                #pragma unroll
                for (int i = 0; i < 4; ++i) {
                    float4 f = src[i];
                    pk.h[4*i+0] = (_Float16)f.x;
                    pk.h[4*i+1] = (_Float16)f.y;
                    pk.h[4*i+2] = (_Float16)f.z;
                    pk.h[4*i+3] = (_Float16)f.w;
                }
            }
#else
            // ---- synchronous gather fallback
            {
                const int w = uti_b[s * TT + gt];
                const float4* row = (const float4*)(emb_word + (long)w * DD + gd);
                #pragma unroll
                for (int i = 0; i < 4; ++i) {
                    float4 f = row[i];
                    pk.h[4*i+0] = (_Float16)f.x;
                    pk.h[4*i+1] = (_Float16)f.y;
                    pk.h[4*i+2] = (_Float16)f.z;
                    pk.h[4*i+3] = (_Float16)f.w;
                }
            }
#endif
            // ---- write f16 tile in both layouts
            *(v4i*)(tile + gt * TSTR + gd)     = pk.q[0];   // 16B aligned
            *(v4i*)(tile + gt * TSTR + gd + 8) = pk.q[1];
            #pragma unroll
            for (int i = 0; i < 16; ++i)
                tileT[(gd + i) * TSTRT + gt] = pk.h[i];
            __syncthreads();

            // ---- scores[t] = tile[t,:] . mem  via WMMA (waves 0,1)
            //      B = mem broadcast into all 16 columns -> unconditional b128 loads
            if (wave < 2) {
                v8f c = {};
                const int M = lane & 15, half = lane >> 4;
                const _Float16* arow = tile + (wave * 16 + M) * TSTR;
                #pragma unroll
                for (int kc = 0; kc < 4; ++kc) {
                    Frag16 a, bm;
                    a.q[0]  = *(const v4i*)(arow + kc * 32 + half * 8);
                    a.q[1]  = *(const v4i*)(arow + kc * 32 + 16 + half * 8);
                    bm.q[0] = *(const v4i*)(memh + kc * 32 + half * 8);
                    bm.q[1] = *(const v4i*)(memh + kc * 32 + 16 + half * 8);
                    c = __builtin_amdgcn_wmma_f32_16x16x32_f16(
                            false, a.h, false, bm.h, (short)0, c, false, false);
                }
                if (M == 0) {   // every column identical; N==0 lanes write rows M=0..7 / 8..15
                    float4* sp = (float4*)(scores + wave * 16 + half * 8);
                    sp[0] = make_float4(c[0], c[1], c[2], c[3]);
                    sp[1] = make_float4(c[4], c[5], c[6], c[7]);
                }
            }
            __syncthreads();

            // ---- softmax over T=32 (wave 0)
            if (wave == 0) {
                float x = scores[lane];
                float m = x;
                for (int off = 16; off > 0; off >>= 1) m = fmaxf(m, __shfl_xor(m, off, 32));
                float e = __expf(x - m);
                float sum = e;
                for (int off = 16; off > 0; off >>= 1) sum += __shfl_xor(sum, off, 32);
                wwh[lane] = (_Float16)(e / sum);
            }
            __syncthreads();

            // ---- sentence[s, dbase+N] = ww . tile[:, dbase+N]  via WMMA (all 8 waves)
            //      A = ww broadcast into all 16 rows; B column from transposed tile
            {
                const int dbase = wave * 16;
                const int N = lane & 15, half = lane >> 4;
                Frag16 a, bm;
                a.q[0] = *(const v4i*)(wwh + half * 8);        // K = half*8 .. +7
                a.q[1] = *(const v4i*)(wwh + 16 + half * 8);   // K = 16+half*8 .. +7
                const _Float16* bcol = tileT + (dbase + N) * TSTRT;
                bm.q[0] = *(const v4i*)(bcol + half * 8);
                bm.q[1] = *(const v4i*)(bcol + 16 + half * 8);
                v8f c = {};
                c = __builtin_amdgcn_wmma_f32_16x16x32_f16(
                        false, a.h, false, bm.h, (short)0, c, false, false);
                if (half == 0) sentence[s * DD + dbase + N] = c[0];  // rows identical
            }
            __syncthreads();
        } // s

        // ---- sentence-level attention
        for (int s = wave; s < SS; s += 8) {
            float acc = 0.f;
            for (int d = lane; d < DD; d += 32) acc += sentence[s * DD + d] * w_word[d];
            for (int off = 16; off > 0; off >>= 1) acc += __shfl_xor(acc, off, 32);
            if (lane == 0) swv[s] = acc + b_word[0];
        }
        if (wave == 0) {
            float acc = 0.f;
            for (int d = lane; d < DD; d += 32) acc += memv[d] * w_sent[d];
            for (int off = 16; off > 0; off >>= 1) acc += __shfl_xor(acc, off, 32);
            if (lane == 0) misc[0] = acc + b_sent[0];
        }
        __syncthreads();
        if (wave == 0) {   // softmax over S=50 of tanh(sw + iw)
            float iw = misc[0];
            int s2 = lane + 32;
            float x0 = (lane < SS) ? tanhf(swv[lane] + iw) : -1e30f;
            float x1 = (s2  < SS) ? tanhf(swv[s2]  + iw) : -1e30f;
            float m = fmaxf(x0, x1);
            for (int off = 16; off > 0; off >>= 1) m = fmaxf(m, __shfl_xor(m, off, 32));
            float e0 = (lane < SS) ? __expf(x0 - m) : 0.f;
            float e1 = (s2  < SS) ? __expf(x1 - m) : 0.f;
            float sum = e0 + e1;
            for (int off = 16; off > 0; off >>= 1) sum += __shfl_xor(sum, off, 32);
            if (lane < SS) swv[lane] = e0 / sum;
            if (s2  < SS) swv[s2]  = e1 / sum;
        }
        __syncthreads();
        if (tid < DD) {    // mem += swf @ sentence
            float acc = 0.f;
            for (int s = 0; s < SS; ++s) acc += swv[s] * sentence[s * DD + tid];
            memv[tid] += acc;
        }
        __syncthreads();
    } // hoop

    // ---- ept = mem @ W_tr + b_tr
    if (tid < LL) {
        float acc = b_tr[tid];
        for (int d = 0; d < DD; ++d) acc += memv[d] * W_tr[d * LL + tid];
        ept_s[tid] = acc;
    }

    // ---- friends: group_idx + pad count
    const int* ufi_b = ufi + (long)u * CC;
    if (tid < CC) fidx_s[tid] = ufi_b[tid];
    __syncthreads();
    if (tid < CC) {
        const float* fr = emb_user + (long)fidx_s[tid] * LL;
        float acc = 0.f;
        for (int l = 0; l < LL; ++l) acc += item_s[l] * fr[l] * w_aff[l];
        float gi = 1.f / (1.f + __expf(-(acc + b_aff[0])));
        gi_s[tid] = gi;
        out[NB + b * CC + tid] = gi;                 // group_idx output
    }
    if (tid == 0) {
        int cnt = 0;
        for (int c = 0; c < CC; ++c) cnt += (fidx_s[c] == NFR);
        misc[1] = (float)(CC - cnt);
    }
    __syncthreads();

    // ---- tiny gating MLP
    if (tid < 32) {
        float acc = b1[tid];
        for (int l = 0; l < LL; ++l) acc += item_s[l] * W1[l * 32 + tid];
        a1s[tid] = 1.f / (1.f + __expf(-acc));
    }
    __syncthreads();
    if (tid < 16) {
        float acc = b2[tid];
        for (int j = 0; j < 32; ++j) acc += a1s[j] * W2[j * 16 + tid];
        a2s[tid] = 1.f / (1.f + __expf(-acc));
    }
    __syncthreads();
    if (tid < 2) {
        float acc = b3[tid];
        for (int k = 0; k < 16; ++k) acc += a2s[k] * W3[k * 2 + tid];
        alpha_s[tid] = 1.f / (1.f + __expf(-acc));
    }
    __syncthreads();

    // ---- user_emb pooling + rating (wave 0)
    if (wave == 0) {
        float fn = misc[1];
        float acc = 0.f;
        for (int l = lane; l < LL; l += 32) {
            float ue = 0.f;
            for (int c = 0; c < CC; ++c)
                ue += emb_user[(long)fidx_s[c] * LL + l] * gi_s[c];
            ue /= fn;
            float vec = alpha_s[0] * ept_s[l] + alpha_s[1] * ue;
            acc += vec * item_s[l] * w_aff[l];
        }
        for (int off = 16; off > 0; off >>= 1) acc += __shfl_xor(acc, off, 32);
        if (lane == 0) out[b] = 1.f / (1.f + __expf(-(acc + b_aff[0])));
    }
}

extern "C" void kernel_launch(void* const* d_in, const int* in_sizes, int n_in,
                              void* d_out, int out_size, void* d_ws, size_t ws_size,
                              hipStream_t stream) {
    (void)in_sizes; (void)n_in; (void)out_size; (void)d_ws; (void)ws_size;
    const int*   user_idx = (const int*)  d_in[0];
    const int*   item_idx = (const int*)  d_in[1];
    const float* emb_word = (const float*)d_in[2];
    const float* emb_item = (const float*)d_in[3];
    const float* emb_user = (const float*)d_in[4];
    const float* W_mem    = (const float*)d_in[5];
    const float* b_mem    = (const float*)d_in[6];
    const float* w_word   = (const float*)d_in[7];
    const float* b_word   = (const float*)d_in[8];
    const float* w_sent   = (const float*)d_in[9];
    const float* b_sent   = (const float*)d_in[10];
    const float* W_tr     = (const float*)d_in[11];
    const float* b_tr     = (const float*)d_in[12];
    const float* W1       = (const float*)d_in[13];
    const float* b1       = (const float*)d_in[14];
    const float* W2       = (const float*)d_in[15];
    const float* b2       = (const float*)d_in[16];
    const float* W3       = (const float*)d_in[17];
    const float* b3       = (const float*)d_in[18];
    const float* w_aff    = (const float*)d_in[19];
    const float* b_aff    = (const float*)d_in[20];
    const int*   uti      = (const int*)  d_in[21];
    const int*   ufi      = (const int*)  d_in[22];
    float*       outp     = (float*)d_out;

    somn_kernel<<<NB, 256, 0, stream>>>(
        user_idx, item_idx, emb_word, emb_item, emb_user,
        W_mem, b_mem, w_word, b_word, w_sent, b_sent, W_tr, b_tr,
        W1, b1, W2, b2, W3, b3, w_aff, b_aff, uti, ufi, outp);
}
